// Stacked_LSTM_36996848288365
// MI455X (gfx1250) — matrix operand order repaired and checked by reading, not compiled, not run on hardware
//
#include <hip/hip_runtime.h>

typedef __attribute__((ext_vector_type(16))) __bf16 v16bf;
typedef __attribute__((ext_vector_type(8)))  float  v8f;

#define T_LEN 2048
#define B_SZ  128
#define F_IN  64
#define U_SZ  128
#define G4U   512

static __device__ __forceinline__ v8f wmma_bf16(v16bf a, v16bf b, v8f c) {
  // D = A(16x32 bf16) x B(32x16 bf16) + C(16x16 f32)
  return __builtin_amdgcn_wmma_f32_16x16x32_bf16(false, a, false, b, (short)0, c,
                                                 false, false);
}

static __device__ __forceinline__ float sigf(float x) {
  return 1.0f / (1.0f + __expf(-x));
}

// ---------------------------------------------------------------------------
// Pack weight [K, 512] fp32 row-major -> bf16 B-matrix tiles.
// B 32x16 bf16 layout (wave32): VGPR v<4: K=2v,2v+1; v>=4: K=2v+8,2v+9; lanes 16-31: +8.
// dst[((kc*32 + nt)*32 + lane)*16 + j]
// ---------------------------------------------------------------------------
__global__ void pack_b_kernel(const float* __restrict__ src, int kTiles,
                              __bf16* __restrict__ dst) {
  int tid = blockIdx.x * blockDim.x + threadIdx.x;
  if (tid >= kTiles * 32 * 32) return;
  int lane = tid & 31;
  int nt   = (tid >> 5) & 31;
  int kc   = tid >> 10;
  int n    = nt * 16 + (lane & 15);
  int khi  = (lane >> 4) * 8;
  __bf16* rec = dst + (size_t)tid * 16;
#pragma unroll
  for (int j = 0; j < 16; ++j) {
    int v = j >> 1;
    int k = kc * 32 + ((v < 4) ? 2 * v : 2 * v + 8) + khi + (j & 1);
    rec[j] = (__bf16)src[(size_t)k * G4U + n];
  }
}

// ---------------------------------------------------------------------------
// Pack x [B, T, F] fp32 -> bf16 A-matrix records: [t][bt][c][lane][16]
// A 16x32 bf16 layout: m = lane%16; VGPR v<4: K=2v(+1); v>=4: K=2v+8;
// lanes 16-31 add +8.
// ---------------------------------------------------------------------------
__global__ void pack_x_kernel(const float* __restrict__ x,
                              __bf16* __restrict__ dst) {
  int tid = blockIdx.x * blockDim.x + threadIdx.x;  // exactly T*8*2*32 threads
  int lane = tid & 31;
  int c    = (tid >> 5) & 1;
  int bt   = (tid >> 6) & 7;
  int t    = tid >> 9;
  int b    = bt * 16 + (lane & 15);
  int khi  = (lane >> 4) * 8;
  __bf16* rec = dst + (size_t)tid * 16;
#pragma unroll
  for (int j = 0; j < 16; ++j) {
    int v   = j >> 1;
    int kin = ((v < 4) ? 2 * v : 2 * v + 8) + khi + (j & 1);
    int f   = c * 32 + kin;
    rec[j]  = (__bf16)x[((size_t)b * T_LEN + t) * F_IN + f];
  }
}

// ---------------------------------------------------------------------------
// Persistent LSTM layer: 8 blocks (one 16-row batch tile each), 8 waves/block
// (one 16-wide u-slice each). Weights live in VGPRs, h exchanged via 4 KB LDS
// in A-pack layout, c-state in registers. Two barriers per timestep.
// x records for step t+1 are prefetched into registers during step t.
// NCX = input K-chunks (F/32). WRITE_SEQ: coalesced h-sequence dump (A-pack).
// ---------------------------------------------------------------------------
template <int NCX, bool WRITE_SEQ, bool WRITE_LAST>
__global__ __launch_bounds__(256) void lstm_layer_kernel(
    const __bf16* __restrict__ xpack, const __bf16* __restrict__ wpack,
    const __bf16* __restrict__ upack, const float* __restrict__ bias,
    __bf16* __restrict__ hseq, float* __restrict__ hlast) {
  __shared__ alignas(32) __bf16 hlds[4 * 32 * 16];  // h in A-pack layout, 4 KB
  const int tidx = threadIdx.x;
  const int lane = tidx & 31;
  const int wave = tidx >> 5;
  const int bt   = blockIdx.x;
  const int lm   = lane & 15;
  const int lhi  = lane >> 4;

  for (int i = tidx; i < 4 * 32 * 16; i += 256) hlds[i] = (__bf16)0.0f;

  // Register-resident weight tiles. Wave w owns N-columns [16w,16w+16) of each
  // gate (gate q -> N-tile index 8q + w).
  const v16bf* wpv = (const v16bf*)wpack;
  const v16bf* upv = (const v16bf*)upack;
  v16bf wX[4][NCX];
  v16bf wU[4][4];
#pragma unroll
  for (int q = 0; q < 4; ++q) {
    const int nt = 8 * q + wave;
#pragma unroll
    for (int c = 0; c < NCX; ++c)
      wX[q][c] = wpv[((size_t)c * 32 + nt) * 32 + lane];
#pragma unroll
    for (int c = 0; c < 4; ++c)
      wU[q][c] = upv[((size_t)c * 32 + nt) * 32 + lane];
  }
  float bq[4];
#pragma unroll
  for (int q = 0; q < 4; ++q) bq[q] = bias[q * U_SZ + wave * 16 + lm];

  // Scatter constants: D-layout element (r,lane) = h(m = r+8*lhi, u = 16w+lm)
  // mapped into A-pack slot (chunk cch, lane lp, byte-pair sj).
  const int sv  = 4 * (wave & 1) + ((lm >> 1) & 3);
  const int shl = (lm >> 3) & 1;
  const int sj  = 2 * sv + (lm & 1);
  const int cch = wave >> 1;

  float cst[8];
#pragma unroll
  for (int r = 0; r < 8; ++r) cst[r] = 0.0f;

  const v16bf* hldsv = (const v16bf*)hlds;
  const v16bf* xrecs = (const v16bf*)xpack;

  // Preload x records for t = 0.
  v16bf ax[NCX];
#pragma unroll
  for (int c = 0; c < NCX; ++c)
    ax[c] = xrecs[((size_t)0 * 8 + bt) * NCX * 32 + c * 32 + lane];

  __syncthreads();

  for (int t = 0; t < T_LEN; ++t) {
    v8f acc[4];
#pragma unroll
    for (int q = 0; q < 4; ++q)
#pragma unroll
      for (int r = 0; r < 8; ++r) acc[q][r] = bq[q];

    // z += x_t @ W  (input projection, fused; then prefetch x_{t+1})
    const int tn = (t + 1 < T_LEN) ? t + 1 : t;
    const v16bf* xnext = xrecs + ((size_t)tn * 8 + bt) * NCX * 32;
#pragma unroll
    for (int c = 0; c < NCX; ++c) {
      v16bf a = ax[c];
#pragma unroll
      for (int q = 0; q < 4; ++q) acc[q] = wmma_bf16(a, wX[q][c], acc[q]);
      ax[c] = xnext[c * 32 + lane];  // overlap global latency with h-part
    }
    // z += h_{t-1} @ U  (recurrent, A from LDS)
#pragma unroll
    for (int c = 0; c < 4; ++c) {
      v16bf ah = hldsv[c * 32 + lane];
#pragma unroll
      for (int q = 0; q < 4; ++q) acc[q] = wmma_bf16(ah, wU[q][c], acc[q]);
    }

    // Gates: i,f,o sigmoid; g relu; c = f*c + i*g; h = o*relu(c)
    float hreg[8];
#pragma unroll
    for (int r = 0; r < 8; ++r) {
      float iv = sigf(acc[0][r]);
      float fv = sigf(acc[1][r]);
      float gv = fmaxf(acc[2][r], 0.0f);
      float ov = sigf(acc[3][r]);
      float cv = fv * cst[r] + iv * gv;
      cst[r]   = cv;
      hreg[r]  = ov * fmaxf(cv, 0.0f);
    }

    __syncthreads();  // all waves finished reading h_{t-1}
#pragma unroll
    for (int r = 0; r < 8; ++r) {
      int m  = r + 8 * lhi;
      int lp = m + 16 * shl;
      hlds[(cch * 32 + lp) * 16 + sj] = (__bf16)hreg[r];
    }
    if (WRITE_LAST && t == T_LEN - 1) {
#pragma unroll
      for (int r = 0; r < 8; ++r) {
        int m = r + 8 * lhi;
        hlast[(size_t)(bt * 16 + m) * U_SZ + wave * 16 + lm] = hreg[r];
      }
    }
    __syncthreads();  // h_t visible to all waves

    if (WRITE_SEQ) {
      // Coalesced A-pack dump of h_t: 256 threads x b128 = 4 KB.
      const int4* src = (const int4*)hlds;
      int4* dst = (int4*)(hseq + ((size_t)t * 8 + bt) * 2048);
      dst[tidx] = src[tidx];
      // In-wave s_wait_dscnt before the store guarantees the LDS reads land
      // before this wave can reach the next iteration's first barrier, which
      // gates the next overwrite of hlds.
    }
  }
}

// ---------------------------------------------------------------------------
// out[b] = h_last[b,:] @ Wd + bd
// ---------------------------------------------------------------------------
__global__ void dense_kernel(const float* __restrict__ hlast,
                             const float* __restrict__ Wd,
                             const float* __restrict__ bd,
                             float* __restrict__ out) {
  int b = threadIdx.x;
  float s = bd[0];
#pragma unroll 8
  for (int u = 0; u < U_SZ; ++u) s += hlast[(size_t)b * U_SZ + u] * Wd[u];
  out[b] = s;
}

extern "C" void kernel_launch(void* const* d_in, const int* in_sizes, int n_in,
                              void* d_out, int out_size, void* d_ws,
                              size_t ws_size, hipStream_t stream) {
  (void)in_sizes; (void)n_in; (void)out_size; (void)ws_size;
  const float* x  = (const float*)d_in[0];
  const float* W1 = (const float*)d_in[1];
  const float* U1 = (const float*)d_in[2];
  const float* b1 = (const float*)d_in[3];
  const float* W2 = (const float*)d_in[4];
  const float* U2 = (const float*)d_in[5];
  const float* b2 = (const float*)d_in[6];
  const float* Wd = (const float*)d_in[7];
  const float* bd = (const float*)d_in[8];
  float* out = (float*)d_out;

  char* ws = (char*)d_ws;
  size_t off = 0;
  __bf16* xpack  = (__bf16*)(ws + off); off += (size_t)T_LEN * 8 * 2 * 512 * 2;  // 32 MB
  __bf16* h1pack = (__bf16*)(ws + off); off += (size_t)T_LEN * 8 * 4 * 512 * 2;  // 64 MB
  __bf16* w1p    = (__bf16*)(ws + off); off += (size_t)2 * 32 * 512 * 2;
  __bf16* u1p    = (__bf16*)(ws + off); off += (size_t)4 * 32 * 512 * 2;
  __bf16* w2p    = (__bf16*)(ws + off); off += (size_t)4 * 32 * 512 * 2;
  __bf16* u2p    = (__bf16*)(ws + off); off += (size_t)4 * 32 * 512 * 2;
  float*  hlast  = (float*)(ws + off);  off += (size_t)B_SZ * U_SZ * 4;

  // Weight packing (B-matrix layout)
  pack_b_kernel<<<(2 * 32 * 32 + 255) / 256, 256, 0, stream>>>(W1, 2, w1p);
  pack_b_kernel<<<(4 * 32 * 32 + 255) / 256, 256, 0, stream>>>(U1, 4, u1p);
  pack_b_kernel<<<(4 * 32 * 32 + 255) / 256, 256, 0, stream>>>(W2, 4, w2p);
  pack_b_kernel<<<(4 * 32 * 32 + 255) / 256, 256, 0, stream>>>(U2, 4, u2p);
  // Input packing (A-matrix layout)
  pack_x_kernel<<<(T_LEN * 8 * 2 * 32) / 256, 256, 0, stream>>>(x, xpack);
  // Layer 1: F=64 -> 2 K-chunks; emits h-sequence in A-pack layout
  lstm_layer_kernel<2, true, false><<<8, 256, 0, stream>>>(
      xpack, w1p, u1p, b1, h1pack, nullptr);
  // Layer 2: F=128 -> 4 K-chunks; emits only final h
  lstm_layer_kernel<4, false, true><<<8, 256, 0, stream>>>(
      h1pack, w2p, u2p, b2, nullptr, hlast);
  // Dense head
  dense_kernel<<<1, 128, 0, stream>>>(hlast, Wd, bd, out);
}
